// Glo_skipAttention_59734405153326
// MI455X (gfx1250) — matrix operand order repaired and checked by reading, not compiled, not run on hardware
//
#include <hip/hip_runtime.h>
#include <hip/hip_bf16.h>
#include <math.h>

typedef __attribute__((ext_vector_type(16))) _Float16 v16h;
typedef __attribute__((ext_vector_type(8)))  _Float16 v8h;
typedef __attribute__((ext_vector_type(8)))  float    v8f;

#define DEVI __device__ __forceinline__

// ---------------- problem constants ----------------
static constexpr int   CCH   = 512;
static constexpr int   HW    = 48 * 48;      // 2304
static constexpr int   NTOK  = 3200;         // 48^2+24^2+16^2+8^2
static constexpr float SCALE = 0.125f;       // 64^-0.5
static constexpr float LNEPS = 1e-5f;

// ---------------- workspace layout (bytes) ----------------
// lifetimes are disjoint where offsets alias (see launch order)
static constexpr size_t OFF_QV      = 0;                 // f32 [2][2][1024][2304]   dead after k_qbuild
static constexpr size_t OFF_GLOB    = 0;                 // f32 [2][2][512][2304]    aliases dead QV
static constexpr size_t OFF_YPACK   = 20ull << 20;       // f16 [2][2] packB(512x2304) aliases dead QV
static constexpr size_t OFF_POOLRAW = 40ull << 20;       // f32 [4][2][512][3200]    dead after k_dwres
static constexpr size_t OFF_KPACK   = 40ull << 20;       // f16 [2][2][8] packB(64x3200) aliases dead POOLRAW
static constexpr size_t OFF_VPACK   = 54ull << 20;       // f16 [2][2][8] packB(3200x64)
static constexpr size_t OFF_Q16     = 68ull << 20;       // f16 [2][2][8][2304][64]
static constexpr size_t OFF_ATT16   = 78ull << 20;       // f16 [2][2][8][2304][64]
static constexpr size_t OFF_POOLOUT = 96ull << 20;       // f32 [4][2][512][3200]
static constexpr size_t OFF_XPACK   = 96ull << 20;       // f16 [2][2] packB(512x2304) dead before POOLOUT written
static constexpr size_t OFF_QVH16   = 150ull << 20;      // f16 1024x512
static constexpr size_t OFF_QVL16   = 151ull << 20;      // f16 1024x512
static constexpr size_t OFF_PW16    = 152ull << 20;      // f16 512x512
static constexpr size_t OFF_WVEC    = (152ull << 20) + (1ull << 19);        // f32 [1024]
static constexpr size_t OFF_WATTN   = (152ull << 20) + (1ull << 19) + 8192; // f16 [8][64][64]

// ---------------- WMMA fragment helpers (CDNA5 wave32 layouts, CDNA5 ISA) --
// packed B-fragment layout: K x N matrix, element (k,n) ->
//   tile=n>>4, kc=k>>5, lane=((k>>4)&1)*16 | (n&15), t=k&15
DEVI size_t packB_idx(int k, int n, int NT) {
  return ((size_t)((k >> 5) * NT + (n >> 4)) * 32 +
          ((((k >> 4) & 1) << 4) | (n & 15))) * 16 + (k & 15);
}

// A fragment (16x32 f16) from row-major source, lda in halves
DEVI v16h load_a_rm(const _Float16* A, int lda, int lane) {
  int m = lane & 15, kb = (lane >> 4) * 16;
  const _Float16* p = A + (size_t)m * lda + kb;
  v8h lo = *(const v8h*)p;
  v8h hi = *(const v8h*)(p + 8);
  v16h a;
#pragma unroll
  for (int i = 0; i < 8; ++i) { a[i] = lo[i]; a[8 + i] = hi[i]; }
  return a;
}

// B fragment (32x16 f16) from row-major K x N source (strided gather)
DEVI v16h load_b_rm(const _Float16* B, int ldb, int lane) {
  int n = lane & 15, kb = (lane >> 4) * 16;
  v16h b;
#pragma unroll
  for (int t = 0; t < 16; ++t) b[t] = B[(size_t)(kb + t) * ldb + n];
  return b;
}

// B fragment from pre-packed layout: 32 contiguous bytes per lane
DEVI v16h load_packed(const _Float16* base, int lane) {
  return *(const v16h*)(base + lane * 16);
}

DEVI v8f wmma_f16(v16h a, v16h b, v8f c) {
  return __builtin_amdgcn_wmma_f32_16x16x32_f16(false, a, false, b, (short)0, c,
                                                false, false);
}

// ---------------- elementwise / conversion kernels ----------------
__global__ void k_f32h(const float* __restrict__ s, _Float16* __restrict__ d, int n) {
  int i = blockIdx.x * 256 + threadIdx.x;
  if (i < n) d[i] = (_Float16)s[i];
}

// pack h_fea / l_fea into WMMA-B layout (K=C rows, N=HW cols)
__global__ void k_xpack(const float* __restrict__ hf, const float* __restrict__ lf,
                        _Float16* __restrict__ xp) {
  size_t i = (size_t)blockIdx.x * 256 + threadIdx.x;
  if (i >= 2ull * 2 * CCH * HW) return;
  int p = (int)(i % HW); size_t t2 = i / HW;
  int c = (int)(t2 % CCH); t2 /= CCH;
  int b = (int)(t2 & 1); int t = (int)(t2 >> 1);
  const float* src = t ? lf : hf;
  float v = src[((size_t)b * CCH + c) * HW + p];
  xp[(size_t)(t * 2 + b) * ((size_t)CCH * HW) + packB_idx(c, p, 144)] = (_Float16)v;
}

// ---------------- QV projection GEMM: [1024x512] x [512x2304] ----------------
// each wave computes a 16x64 tile: one A fragment feeds 4 accumulators
__global__ void __launch_bounds__(128) k_gemm_qv(const _Float16* __restrict__ wh,
                                                 const _Float16* __restrict__ wl,
                                                 const _Float16* __restrict__ xpack,
                                                 float* __restrict__ qv) {
  int wid = (blockIdx.x * blockDim.x + threadIdx.x) >> 5;
  int lane = threadIdx.x & 31;
  // wid = ((t*2+b)*64 + ot)*36 + pg   (pg: group of 4 consecutive 16-col tiles)
  int pg = wid % 36; int tmp = wid / 36;
  int ot = tmp % 64; tmp /= 64;
  int b = tmp & 1; int t = tmp >> 1;
  const _Float16* W = t ? wl : wh;
  const _Float16* X = xpack + (size_t)(t * 2 + b) * ((size_t)CCH * HW);
  v8f a0 = {}, a1 = {}, a2 = {}, a3 = {};
#pragma unroll 2
  for (int kc = 0; kc < 16; ++kc) {
    v16h af = load_a_rm(W + (size_t)(ot * 16) * 512 + kc * 32, 512, lane);
    const _Float16* Xk = X + (size_t)(kc * 144 + pg * 4) * 512;
    v16h b0 = load_packed(Xk, lane);
    v16h b1 = load_packed(Xk + 512, lane);
    v16h b2 = load_packed(Xk + 1024, lane);
    v16h b3 = load_packed(Xk + 1536, lane);
    a0 = wmma_f16(af, b0, a0); a1 = wmma_f16(af, b1, a1);
    a2 = wmma_f16(af, b2, a2); a3 = wmma_f16(af, b3, a3);
  }
  int o = ot * 16 + (lane >> 4) * 8;
  int p = pg * 64 + (lane & 15);
  float* out = qv + (size_t)(t * 2 + b) * 1024 * HW;
#pragma unroll
  for (int r = 0; r < 8; ++r) {
    float* orow = out + (size_t)(o + r) * HW + p;
    orow[0] = a0[r]; orow[16] = a1[r]; orow[32] = a2[r]; orow[48] = a3[r];
  }
}

// ---------------- multi-ratio average pooling ----------------
__global__ void k_pool(const float* __restrict__ qv, float* __restrict__ praw) {
  size_t i = (size_t)blockIdx.x * 256 + threadIdx.x;
  if (i >= 4ull * 2 * CCH * NTOK) return;
  int tok = (int)(i % NTOK); size_t t2 = i / NTOK;
  int c = (int)(t2 % CCH); t2 /= CCH;
  int b = (int)(t2 & 1); int ten = (int)(t2 >> 1);
  int oh, loc;
  if (tok < 2304)      { oh = 48; loc = tok; }
  else if (tok < 2880) { oh = 24; loc = tok - 2304; }
  else if (tok < 3136) { oh = 16; loc = tok - 2880; }
  else                 { oh = 8;  loc = tok - 3136; }
  int r = 48 / oh;
  int py = loc / oh, px = loc - py * oh;
  int o = ((ten & 1) << 9) + c;
  const float* row = qv + ((size_t)((ten >> 1) * 2 + b) * 1024 + o) * HW;
  float sum = 0.f;
  for (int dy = 0; dy < r; ++dy)
    for (int dx = 0; dx < r; ++dx)
      sum += row[(py * r + dy) * 48 + px * r + dx];
  praw[i] = sum / (float)(r * r);
}

// ---------------- depthwise 3x3 + bias + residual on pooled maps ------------
__global__ void k_dwres(const float* __restrict__ praw, const float* __restrict__ dw,
                        const float* __restrict__ db, float* __restrict__ pout) {
  size_t i = (size_t)blockIdx.x * 256 + threadIdx.x;
  if (i >= 4ull * 2 * CCH * NTOK) return;
  int tok = (int)(i % NTOK); size_t t2 = i / NTOK;
  int c = (int)(t2 % CCH); t2 /= CCH;
  int ten = (int)(t2 >> 1);
  int s, oh, base;
  if (tok < 2304)      { s = 0; oh = 48; base = 0; }
  else if (tok < 2880) { s = 1; oh = 24; base = 2304; }
  else if (tok < 3136) { s = 2; oh = 16; base = 2880; }
  else                 { s = 3; oh = 8;  base = 3136; }
  int loc = tok - base;
  int py = loc / oh, px = loc - py * oh;
  const float* rowp = praw + (i - tok) + base;   // start of this (ten,b,c) section
  const float* w = dw + ((size_t)s * CCH + c) * 9;
  float acc = db[s * CCH + c];
#pragma unroll
  for (int ky = 0; ky < 3; ++ky) {
    int yy = py + ky - 1;
    if (yy < 0 || yy >= oh) continue;
#pragma unroll
    for (int kx = 0; kx < 3; ++kx) {
      int xx = px + kx - 1;
      if (xx < 0 || xx >= oh) continue;
      acc += w[ky * 3 + kx] * rowp[yy * oh + xx];
    }
  }
  pout[i] = praw[i] + acc;
}

// ---------------- LayerNorm over C, write WMMA-packed K / V ----------------
__global__ void __launch_bounds__(128) k_ln(const float* __restrict__ pout,
                                            const float* __restrict__ g,
                                            const float* __restrict__ be,
                                            _Float16* __restrict__ kpack,
                                            _Float16* __restrict__ vpack) {
  int wid = (blockIdx.x * blockDim.x + threadIdx.x) >> 5;
  int lane = threadIdx.x & 31;
  int tok = wid % NTOK; int tb = wid / NTOK;
  int b = tb & 1; int ten = tb >> 1;
  const float* rowbase = pout + (size_t)tb * CCH * NTOK + tok;
  float xv[16]; float s = 0.f, sq = 0.f;
#pragma unroll
  for (int i = 0; i < 16; ++i) {
    float v = rowbase[(size_t)(lane + 32 * i) * NTOK];
    xv[i] = v; s += v; sq += v * v;
  }
#pragma unroll
  for (int m = 16; m >= 1; m >>= 1) { s += __shfl_xor(s, m); sq += __shfl_xor(sq, m); }
  float mu = s * (1.f / 512.f);
  float var = sq * (1.f / 512.f) - mu * mu;
  float inv = rsqrtf(var + LNEPS);
  int pr = ten >> 1;            // attention pair
  bool isV = (ten & 1) != 0;
#pragma unroll
  for (int i = 0; i < 16; ++i) {
    int c = lane + 32 * i;
    float y = (xv[i] - mu) * inv * g[c] + be[c];
    int head = c >> 6, dh = c & 63;
    size_t basehb = (size_t)((pr * 2 + b) * 8 + head) * (64ull * NTOK);
    if (!isV) kpack[basehb + packB_idx(dh, tok, 200)] = (_Float16)y;   // B of Q*K^T
    else      vpack[basehb + packB_idx(tok, dh, 4)]   = (_Float16)y;   // B of P*V
  }
}

// ---------------- build f16 Q (scale folded in) ----------------
__global__ void k_qbuild(const float* __restrict__ qv, _Float16* __restrict__ q16) {
  size_t i = (size_t)blockIdx.x * 256 + threadIdx.x;
  if (i >= 2ull * 2 * 8 * HW * 64) return;
  int dh = (int)(i & 63); size_t t2 = i >> 6;
  int sp = (int)(t2 % HW); t2 /= HW;
  int h = (int)(t2 & 7); t2 >>= 3;
  int b = (int)(t2 & 1); int qt = (int)(t2 >> 1);
  int src_t = qt ? 0 : 1;      // attend0 queries come from l, attend1 from h
  float v = qv[((size_t)(src_t * 2 + b) * 1024 + (dh * 8 + h)) * HW + sp] * SCALE;
  q16[i] = (_Float16)v;
}

// ---------------- flash attention: 16 q-rows x 3200 keys per wave ----------
__global__ void __launch_bounds__(128) k_attn(const _Float16* __restrict__ q16,
                                              const _Float16* __restrict__ kpack,
                                              const _Float16* __restrict__ vpack,
                                              _Float16* __restrict__ att16) {
  __shared__ _Float16 sP[4 * 512];
  int wslot = threadIdx.x >> 5, lane = threadIdx.x & 31;
  int wid = blockIdx.x * 4 + wslot;
  int it = wid % 144; int abh = wid / 144;
  const _Float16* Q  = q16   + ((size_t)abh * HW + it * 16) * 64;
  const _Float16* Kb = kpack + (size_t)abh * (64ull * NTOK);
  const _Float16* Vb = vpack + (size_t)abh * (64ull * NTOK);
  v16h A0 = load_a_rm(Q, 64, lane);        // dh 0..31
  v16h A1 = load_a_rm(Q + 32, 64, lane);   // dh 32..63
  float mr[8], lr[8];
  v8f O0 = {}, O1 = {}, O2 = {}, O3 = {};
#pragma unroll
  for (int r = 0; r < 8; ++r) { mr[r] = -3.0e38f; lr[r] = 0.f; }
  _Float16* myP = sP + wslot * 512;
  for (int nt = 0; nt < 100; ++nt) {       // 32 keys per step
    v16h B00 = load_packed(Kb + (size_t)((0 * 200 + 2 * nt) * 32) * 16, lane);
    v16h B01 = load_packed(Kb + (size_t)((1 * 200 + 2 * nt) * 32) * 16, lane);
    v16h B10 = load_packed(Kb + (size_t)((0 * 200 + 2 * nt + 1) * 32) * 16, lane);
    v16h B11 = load_packed(Kb + (size_t)((1 * 200 + 2 * nt + 1) * 32) * 16, lane);
    v8f s0 = {}; s0 = wmma_f16(A0, B00, s0); s0 = wmma_f16(A1, B01, s0);
    v8f s1 = {}; s1 = wmma_f16(A0, B10, s1); s1 = wmma_f16(A1, B11, s1);
    float e0[8], e1[8];
#pragma unroll
    for (int r = 0; r < 8; ++r) {
      float t = fmaxf(s0[r], s1[r]);
      t = fmaxf(t, __shfl_xor(t, 1)); t = fmaxf(t, __shfl_xor(t, 2));
      t = fmaxf(t, __shfl_xor(t, 4)); t = fmaxf(t, __shfl_xor(t, 8));
      float mn = fmaxf(mr[r], t);
      float al = __expf(mr[r] - mn); mr[r] = mn;
      float a0 = __expf(s0[r] - mn), a1 = __expf(s1[r] - mn);
      e0[r] = a0; e1[r] = a1;
      float rs = a0 + a1;
      rs += __shfl_xor(rs, 1); rs += __shfl_xor(rs, 2);
      rs += __shfl_xor(rs, 4); rs += __shfl_xor(rs, 8);
      lr[r] = lr[r] * al + rs;
      O0[r] *= al; O1[r] *= al; O2[r] *= al; O3[r] *= al;
    }
    // C-fragment -> A-fragment transpose through LDS
    int row0 = (lane >> 4) * 8, col = lane & 15;
#pragma unroll
    for (int r = 0; r < 8; ++r) {
      myP[(row0 + r) * 32 + col]      = (_Float16)e0[r];
      myP[(row0 + r) * 32 + 16 + col] = (_Float16)e1[r];
    }
    __syncthreads();
    v16h Pa;
    int mm = lane & 15, kb2 = (lane >> 4) * 16;
#pragma unroll
    for (int i2 = 0; i2 < 8; ++i2) {
      Pa[i2]     = myP[mm * 32 + kb2 + i2];
      Pa[8 + i2] = myP[mm * 32 + kb2 + 8 + i2];
    }
    __syncthreads();
    v16h V0 = load_packed(Vb + (size_t)((nt * 4 + 0) * 32) * 16, lane);
    v16h V1 = load_packed(Vb + (size_t)((nt * 4 + 1) * 32) * 16, lane);
    v16h V2 = load_packed(Vb + (size_t)((nt * 4 + 2) * 32) * 16, lane);
    v16h V3 = load_packed(Vb + (size_t)((nt * 4 + 3) * 32) * 16, lane);
    O0 = wmma_f16(Pa, V0, O0); O1 = wmma_f16(Pa, V1, O1);
    O2 = wmma_f16(Pa, V2, O2); O3 = wmma_f16(Pa, V3, O3);
  }
  int col = lane & 15, rbase = (lane >> 4) * 8;
  _Float16* outb = att16 + ((size_t)abh * HW + it * 16) * 64;
#pragma unroll
  for (int r = 0; r < 8; ++r) {
    float inv = 1.0f / lr[r];
    int row = rbase + r;
    outb[(size_t)row * 64 + 0  + col] = (_Float16)(O0[r] * inv);
    outb[(size_t)row * 64 + 16 + col] = (_Float16)(O1[r] * inv);
    outb[(size_t)row * 64 + 32 + col] = (_Float16)(O2[r] * inv);
    outb[(size_t)row * 64 + 48 + col] = (_Float16)(O3[r] * inv);
  }
}

// ---------------- glob_branch constants: wvec = gelu(ln1_b)*rowsum(qk_w)+qk_b
__global__ void k_wvec(const float* __restrict__ qkw, const float* __restrict__ qkb,
                       const float* __restrict__ ln1b, float* __restrict__ wvec) {
  int o = blockIdx.x * 256 + threadIdx.x;
  if (o >= 1024) return;
  float s = 0.f;
  for (int c = 0; c < 512; ++c) s += qkw[o * 512 + c];
  float x = ln1b[0];
  float ge = 0.5f * x * (1.0f + erff(x * 0.70710678f));
  wvec[o] = ge * s + qkb[o];
}

__global__ void k_wattn(const float* __restrict__ wvec, _Float16* __restrict__ wat) {
  __shared__ float sl[64];
  int h = blockIdx.x >> 6, i = blockIdx.x & 63, j = threadIdx.x;
  float wq = wvec[h * 128 + i];
  float wk = wvec[h * 128 + 64 + j];
  float lg = SCALE * wq * wk;
  sl[j] = lg;
  __syncthreads();
  float mx = -3.0e38f;
  for (int t = 0; t < 64; ++t) mx = fmaxf(mx, sl[t]);
  float sm = 0.f;
  for (int t = 0; t < 64; ++t) sm += __expf(sl[t] - mx);
  wat[((size_t)h * 64 + i) * 64 + j] = (_Float16)(__expf(lg - mx) / sm);
}

// ---------------- agg GEMM: att_fea(2304x64) x w_attn(64x64) per head -------
__global__ void __launch_bounds__(128) k_agg(const _Float16* __restrict__ att16,
                                             const _Float16* __restrict__ wat,
                                             float* __restrict__ glob) {
  int wid = (blockIdx.x * blockDim.x + threadIdx.x) >> 5;
  int lane = threadIdx.x & 31;
  int it = wid % 144; int abh = wid / 144;
  int h = abh & 7; int ab = abh >> 3;
  const _Float16* A = att16 + ((size_t)abh * HW + it * 16) * 64;
  v16h A0 = load_a_rm(A, 64, lane);
  v16h A1 = load_a_rm(A + 32, 64, lane);
  const _Float16* W = wat + (size_t)h * 4096;
#pragma unroll
  for (int ntile = 0; ntile < 4; ++ntile) {
    v16h Bf0 = load_b_rm(W + ntile * 16, 64, lane);            // i 0..31
    v16h Bf1 = load_b_rm(W + 32 * 64 + ntile * 16, 64, lane);  // i 32..63
    v8f acc = {};
    acc = wmma_f16(A0, Bf0, acc);
    acc = wmma_f16(A1, Bf1, acc);
    int w = ntile * 16 + (lane & 15);
    int cch = w * 8 + h;                 // c = dh*HEADS + head
    int p0 = it * 16 + (lane >> 4) * 8;
    float* outp = glob + ((size_t)ab * CCH + cch) * HW;
#pragma unroll
    for (int r = 0; r < 8; ++r) outp[p0 + r] = acc[r];
  }
}

// ---------------- final depthwise 3x3 -> packed f16 ----------------
__global__ void k_dwc_out(const float* __restrict__ glob, const float* __restrict__ dw,
                          _Float16* __restrict__ ypack) {
  size_t i = (size_t)blockIdx.x * 256 + threadIdx.x;
  if (i >= 2ull * 2 * CCH * HW) return;
  int p = (int)(i % HW); size_t t2 = i / HW;
  int c = (int)(t2 % CCH); int ab = (int)(t2 / CCH);
  int py = p / 48, px = p - py * 48;
  const float* src = glob + ((size_t)ab * CCH + c) * HW;
  const float* w = dw + (size_t)c * 9;
  float acc = 0.f;
#pragma unroll
  for (int ky = 0; ky < 3; ++ky) {
    int yy = py + ky - 1;
    if (yy < 0 || yy >= 48) continue;
#pragma unroll
    for (int kx = 0; kx < 3; ++kx) {
      int xx = px + kx - 1;
      if (xx < 0 || xx >= 48) continue;
      acc += w[ky * 3 + kx] * src[yy * 48 + xx];
    }
  }
  ypack[(size_t)ab * ((size_t)CCH * HW) + packB_idx(c, p, 144)] = (_Float16)acc;
}

// ---------------- final 1x1 conv GEMM: [512x512] x [512x2304] ----------------
// 16x64 tile per wave, A fragment reused across 4 accumulators
__global__ void __launch_bounds__(128) k_gemm_out(const _Float16* __restrict__ pw16,
                                                  const _Float16* __restrict__ ypack,
                                                  float* __restrict__ out) {
  int wid = (blockIdx.x * blockDim.x + threadIdx.x) >> 5;
  int lane = threadIdx.x & 31;
  // wid = (ab*32 + ot)*36 + pg
  int pg = wid % 36; int tmp = wid / 36;
  int ot = tmp & 31; int ab = tmp >> 5;
  const _Float16* Y = ypack + (size_t)ab * ((size_t)CCH * HW);
  v8f a0 = {}, a1 = {}, a2 = {}, a3 = {};
#pragma unroll 2
  for (int kc = 0; kc < 16; ++kc) {
    v16h af = load_a_rm(pw16 + (size_t)(ot * 16) * 512 + kc * 32, 512, lane);
    const _Float16* Yk = Y + (size_t)(kc * 144 + pg * 4) * 512;
    v16h b0 = load_packed(Yk, lane);
    v16h b1 = load_packed(Yk + 512, lane);
    v16h b2 = load_packed(Yk + 1024, lane);
    v16h b3 = load_packed(Yk + 1536, lane);
    a0 = wmma_f16(af, b0, a0); a1 = wmma_f16(af, b1, a1);
    a2 = wmma_f16(af, b2, a2); a3 = wmma_f16(af, b3, a3);
  }
  int o = ot * 16 + (lane >> 4) * 8;
  int p = pg * 64 + (lane & 15);
  float* op = out + (size_t)ab * CCH * HW;
#pragma unroll
  for (int r = 0; r < 8; ++r) {
    float* orow = op + (size_t)(o + r) * HW + p;
    orow[0] = a0[r]; orow[16] = a1[r]; orow[32] = a2[r]; orow[48] = a3[r];
  }
}

// ---------------- host side ----------------
extern "C" void kernel_launch(void* const* d_in, const int* in_sizes, int n_in,
                              void* d_out, int out_size, void* d_ws, size_t ws_size,
                              hipStream_t stream) {
  const float* h_fea   = (const float*)d_in[0];
  const float* l_fea   = (const float*)d_in[1];
  const float* qvh_w   = (const float*)d_in[2];
  const float* qvl_w   = (const float*)d_in[3];
  const float* dconv_w = (const float*)d_in[4];
  const float* dconv_b = (const float*)d_in[5];
  const float* norm_g  = (const float*)d_in[6];
  const float* norm_b  = (const float*)d_in[7];
  const float* ln1_b   = (const float*)d_in[11];
  const float* qk_w    = (const float*)d_in[12];
  const float* qk_b    = (const float*)d_in[13];
  const float* outh_dw = (const float*)d_in[14];
  const float* outh_pw = (const float*)d_in[15];

  char* ws = (char*)d_ws;
  float*     QV    = (float*)(ws + OFF_QV);
  float*     GLOB  = (float*)(ws + OFF_GLOB);
  _Float16*  YPACK = (_Float16*)(ws + OFF_YPACK);
  float*     PRAW  = (float*)(ws + OFF_POOLRAW);
  _Float16*  KPACK = (_Float16*)(ws + OFF_KPACK);
  _Float16*  VPACK = (_Float16*)(ws + OFF_VPACK);
  _Float16*  Q16   = (_Float16*)(ws + OFF_Q16);
  _Float16*  ATT16 = (_Float16*)(ws + OFF_ATT16);
  float*     POUT  = (float*)(ws + OFF_POOLOUT);
  _Float16*  XPACK = (_Float16*)(ws + OFF_XPACK);
  _Float16*  W16H  = (_Float16*)(ws + OFF_QVH16);
  _Float16*  W16L  = (_Float16*)(ws + OFF_QVL16);
  _Float16*  PW16  = (_Float16*)(ws + OFF_PW16);
  float*     WVEC  = (float*)(ws + OFF_WVEC);
  _Float16*  WATTN = (_Float16*)(ws + OFF_WATTN);

  // weight conversions
  k_f32h<<<(1024 * 512 + 255) / 256, 256, 0, stream>>>(qvh_w, W16H, 1024 * 512);
  k_f32h<<<(1024 * 512 + 255) / 256, 256, 0, stream>>>(qvl_w, W16L, 1024 * 512);
  k_f32h<<<(512 * 512 + 255) / 256, 256, 0, stream>>>(outh_pw, PW16, 512 * 512);
  // pack inputs for WMMA
  k_xpack<<<36864, 256, 0, stream>>>(h_fea, l_fea, XPACK);
  // QV projection (both branches, both batches): 2*2*64*36 = 9216 waves
  k_gemm_qv<<<2304, 128, 0, stream>>>(W16H, W16L, XPACK, QV);
  // pooling + dwconv residual
  k_pool<<<51200, 256, 0, stream>>>(QV, PRAW);
  k_dwres<<<51200, 256, 0, stream>>>(PRAW, dconv_w, dconv_b, POUT);
  // layernorm -> packed K/V
  k_ln<<<6400, 128, 0, stream>>>(POUT, norm_g, norm_b, KPACK, VPACK);
  // queries (scale folded)
  k_qbuild<<<18432, 256, 0, stream>>>(QV, Q16);
  // flash attention, both branches
  k_attn<<<2304, 128, 0, stream>>>(Q16, KPACK, VPACK, ATT16);
  // constant w_attn of glob_branch
  k_wvec<<<4, 256, 0, stream>>>(qk_w, qk_b, ln1_b, WVEC);
  k_wattn<<<512, 64, 0, stream>>>(WVEC, WATTN);
  // per-head aggregation GEMM
  k_agg<<<2304, 128, 0, stream>>>(ATT16, WATTN, GLOB);
  // output projection: depthwise 3x3 then 1x1 conv
  k_dwc_out<<<18432, 256, 0, stream>>>(GLOB, outh_dw, YPACK);
  // final GEMM: 2*2*32*36 = 4608 waves
  k_gemm_out<<<1152, 128, 0, stream>>>(PW16, YPACK, (float*)d_out);
}
